// BahdanauAttention_13907104104900
// MI455X (gfx1250) — matrix operand order repaired and checked by reading, not compiled, not run on hardware
//
#include <hip/hip_runtime.h>
#include <math.h>

#define H 1024
#define S 2048
#define B 32
#define TM 128            // keys rows per WG in score kernel
#define AST 1032          // Als row stride in bf16 elems (2064B = 516 DW: conflict-free)
#define BROWB 2080        // Bls row stride in bytes (2048 + 2x16B TDM pad = 520 DW)

typedef __attribute__((ext_vector_type(16))) __bf16 v16bf;
typedef __attribute__((ext_vector_type(8)))  __bf16 v8bf;
typedef __attribute__((ext_vector_type(8)))  float  v8f;
typedef __attribute__((ext_vector_type(4)))  unsigned int u32x4;
typedef __attribute__((ext_vector_type(8)))  int i32x8;
typedef __attribute__((ext_vector_type(4)))  int i32x4;

// ---------------------------------------------------------------------------
// Kernel 0: Ua16T[n*H + k] = bf16(Ua_w[k*H + n])  (transpose + convert)
// Transposed layout => every 16-column WMMA B-panel is one CONTIGUOUS 32KB
// block -> ideal for a single TDM descriptor per panel.
// ---------------------------------------------------------------------------
__global__ __launch_bounds__(256)
void cvt_transpose_kernel(const float* __restrict__ Ua_w,
                          unsigned short* __restrict__ ua16_)
{
    __bf16* Ua16T = reinterpret_cast<__bf16*>(ua16_);
    int t = blockIdx.x * 256 + threadIdx.x;     // 0 .. H*H-1
    int n = t >> 10;
    int k = t & (H - 1);
    Ua16T[t] = (__bf16)Ua_w[(size_t)k * H + n];
}

// ---------------------------------------------------------------------------
// Kernel 1: qp[b*H + d] = sum_h query[b*H+h] * Wa_w[h*H+d] + Wa_b[d]
// ---------------------------------------------------------------------------
__global__ __launch_bounds__(256)
void qproj_kernel(const float* __restrict__ query,
                  const float* __restrict__ Wa_w,
                  const float* __restrict__ Wa_b,
                  float* __restrict__ qp)
{
    int t = blockIdx.x * 256 + threadIdx.x;     // 0 .. B*H-1
    int b = t >> 10;
    int d = t & (H - 1);
    float acc = Wa_b[d];
    const float* qrow = query + (size_t)b * H;
    for (int h = 0; h < H; ++h)
        acc = fmaf(qrow[h], Wa_w[(size_t)h * H + d], acc);
    qp[t] = acc;
}

// ---------------------------------------------------------------------------
// TDM: copy one contiguous 16-col B panel (16 x 1024 bf16 = 32KB) from global
// to LDS, inserting 16B padding per 1KB (pad_interval=7 -> 256 DW,
// pad_amount=3 -> 4 DW) so the LDS panel has a 2080B row stride.
// 2D descriptor: data_size=2B, tile_dim0=1024, tile_dim1=16, dim0_stride=1024.
// This toolchain exposes the 6-arg builtin: (g0, g1, g2, g3, g4, cpol).
// ---------------------------------------------------------------------------
__device__ __forceinline__ void tdm_load_b_panel(const unsigned short* src,
                                                 unsigned lds_byte_addr)
{
    unsigned long long ga = (unsigned long long)(size_t)src;
    u32x4 g0;
    g0.x = 1u;                                        // count=1, user mode
    g0.y = lds_byte_addr;                             // lds_addr
    g0.z = (unsigned)(ga & 0xffffffffull);            // global_addr[31:0]
    g0.w = (unsigned)((ga >> 32) & 0x01ffffffull)     // global_addr[56:32]
         | 0x80000000u;                               // type=2 ("image")
    i32x8 g1;
    g1[0] = (int)((1u << 16)        // data_size = 2 bytes
                | (1u << 20)        // pad_enable
                | (7u << 22)        // pad_interval: 256 DW = 1KB
                | (3u << 25));      // pad_amount: 4 DW = 16B
    g1[1] = (int)(1024u << 16);     // tensor_dim0[15:0] = 1024
    g1[2] = (int)(1024u << 16);     // tensor_dim0 hi=0 | tensor_dim1[15:0]=1024
    g1[3] = (int)(1024u << 16);     // tensor_dim1 hi=0 | tile_dim0=1024
    g1[4] = 16;                     // tile_dim1 = 16 rows, tile_dim2 = 0
    g1[5] = 1024;                   // tensor_dim0_stride = 1024 (elems)
    g1[6] = 0;
    g1[7] = 0;
    i32x4 z4 = {0, 0, 0, 0};
    i32x8 z8 = {0, 0, 0, 0, 0, 0, 0, 0};
    __builtin_amdgcn_tensor_load_to_lds(g0, g1, z4, z4, z8, 0);
}

// ---------------------------------------------------------------------------
// Kernel 2: fused scores[r] = sum_d tanh(keys[r,:]·Ua_w[:,d] + qp[b,d] + Ua_b[d]) * Va_w[d] + Va_b
// One WG = TM=128 keys rows (lies inside one b since S % 128 == 0).
// 8 waves, wave w owns rows [16w, 16w+16). A-tile (128x1024 bf16) lives in
// LDS for the whole kernel; B panels (16 cols x K) are TDM-streamed into LDS
// -> L2 B-traffic drops to 512 WG x 2MB = 1 GB (vs 8 GB at TM=16).
// 4 independent K-chains per wave for WMMA ILP.
// ---------------------------------------------------------------------------
__global__ __launch_bounds__(256)
void score_kernel(const float* __restrict__ keys,
                  const unsigned short* __restrict__ ua16_,
                  const float* __restrict__ qp,
                  const float* __restrict__ Ua_b,
                  const float* __restrict__ Va_w,
                  const float* __restrict__ Va_b,
                  float* __restrict__ scores)
{
    __shared__ __bf16 Als[TM * AST];                // ~258 KB
    __shared__ __bf16 Bls[16 * (BROWB / 2)];        // ~32.5 KB (TDM-padded rows)

    const int rowbase = blockIdx.x * TM;            // row index into [B*S]
    const int b       = rowbase >> 11;              // rowbase / S

    // Stage A tile: coalesced fp32 read of keys, convert to bf16.
    for (int idx = threadIdx.x; idx < TM * H; idx += 256) {
        int r = idx >> 10;
        int k = idx & (H - 1);
        Als[r * AST + k] = (__bf16)keys[(size_t)(rowbase + r) * H + k];
    }
    __syncthreads();

    const int lane = threadIdx.x & 31;
    const int wave = threadIdx.x >> 5;
    const int hi   = lane >> 4;                     // 0 or 1
    const int l15  = lane & 15;
    const int mrow = wave << 4;                     // this wave's row sub-tile
    const unsigned bls_base = (unsigned)(size_t)(void*)&Bls[0];

    float racc[8];
    #pragma unroll
    for (int j = 0; j < 8; ++j) racc[j] = 0.0f;
    const float vb = Va_b[0];

    for (int nt = 0; nt < 64; ++nt) {
        if (wave == 0) {
            tdm_load_b_panel(ua16_ + (size_t)nt * 16 * H, bls_base);
            __builtin_amdgcn_s_wait_tensorcnt(0);
        }
        __syncthreads();                            // B panel ready

        v8f cacc[4];
        #pragma unroll
        for (int c = 0; c < 4; ++c)
            #pragma unroll
            for (int j = 0; j < 8; ++j) cacc[c][j] = 0.0f;

        for (int k0 = 0; k0 < H; k0 += 128) {
            #pragma unroll
            for (int c = 0; c < 4; ++c) {           // 4 independent chains
                const int kk = k0 + (c << 5);
                // A 16x32 k order used here (same as B): lanes<16: M=lane, K 0..15;
                //                      lanes>=16: M=lane-16, K 16..31
                const __bf16* arow = &Als[(mrow + l15) * AST + kk + (hi << 4)];
                v8bf alo = *(const v8bf*)(arow);
                v8bf ahi = *(const v8bf*)(arow + 8);
                v16bf a = __builtin_shufflevector(alo, ahi,
                            0,1,2,3,4,5,6,7,8,9,10,11,12,13,14,15);
                // B 32x16 k order used here (same as A): lanes<16: N=lane, K=0..15; lanes>=16: K=16..31
                const int kbase = (hi << 4) + kk;
                const char* bp = (const char*)Bls + l15 * BROWB
                               + kbase * 2 + ((kbase >> 9) << 4); // skip TDM pads
                v8bf blo = *(const v8bf*)(bp);
                v8bf bhi = *(const v8bf*)(bp + 16);
                v16bf bm = __builtin_shufflevector(blo, bhi,
                            0,1,2,3,4,5,6,7,8,9,10,11,12,13,14,15);
                cacc[c] = __builtin_amdgcn_wmma_f32_16x16x32_bf16(
                              false, a, false, bm, (short)0, cacc[c], false, false);
            }
        }

        // Epilogue for this 16-col panel. C layout: vgpr j -> M = j + 8*hi, N = l15
        const int col  = (nt << 4) + l15;
        const float qv = qp[(size_t)b * H + col] + Ua_b[col];
        const float va = Va_w[col];
        #pragma unroll
        for (int j = 0; j < 8; ++j) {
            float t = (cacc[0][j] + cacc[1][j]) + (cacc[2][j] + cacc[3][j]);
            racc[j] += tanhf(t + qv) * va;
        }
        __syncthreads();                            // done reading Bls
    }

    // Reduce over the 16 columns held across each 16-lane half.
    #pragma unroll
    for (int j = 0; j < 8; ++j) {
        float v = racc[j];
        v += __shfl_xor(v, 1, 32);
        v += __shfl_xor(v, 2, 32);
        v += __shfl_xor(v, 4, 32);
        v += __shfl_xor(v, 8, 32);
        racc[j] = v;
    }
    if (l15 == 0) {                                 // lanes 0 and 16
        const int r0 = rowbase + mrow + (hi << 3);
        #pragma unroll
        for (int j = 0; j < 8; ++j)
            scores[r0 + j] = racc[j] + vb;
    }
}

// ---------------------------------------------------------------------------
// Kernel 3: softmax over S per batch; writes weights to d_out[B*H ...]
// ---------------------------------------------------------------------------
__global__ __launch_bounds__(256)
void softmax_kernel(const float* __restrict__ scores,
                    float* __restrict__ wout)
{
    const int b = blockIdx.x;
    const int t = threadIdx.x;
    __shared__ float red[256];
    const float* row = scores + (size_t)b * S;

    float m = -3.402823466e+38f;
    for (int s = t; s < S; s += 256) m = fmaxf(m, row[s]);
    red[t] = m; __syncthreads();
    for (int off = 128; off > 0; off >>= 1) {
        if (t < off) red[t] = fmaxf(red[t], red[t + off]);
        __syncthreads();
    }
    m = red[0]; __syncthreads();

    float sum = 0.0f;
    for (int s = t; s < S; s += 256) sum += __expf(row[s] - m);
    red[t] = sum; __syncthreads();
    for (int off = 128; off > 0; off >>= 1) {
        if (t < off) red[t] += red[t + off];
        __syncthreads();
    }
    const float inv = 1.0f / red[0];

    for (int s = t; s < S; s += 256)
        wout[(size_t)b * S + s] = __expf(row[s] - m) * inv;
}

// ---------------------------------------------------------------------------
// Kernel 4: context[b,h] = sum_s w[b,s] * keys[b,s,h]   (pure HBM bandwidth)
// ---------------------------------------------------------------------------
__global__ __launch_bounds__(256)
void context_kernel(const float* __restrict__ keys,
                    const float* __restrict__ w,
                    float* __restrict__ ctx)
{
    const int b = blockIdx.x >> 2;                 // H/256 = 4 blocks per b
    const int h = ((blockIdx.x & 3) << 8) + threadIdx.x;
    const float* kb = keys + (size_t)b * S * H + h;
    const float* wb = w + (size_t)b * S;
    float acc = 0.0f;
    for (int s = 0; s < S; ++s)
        acc = fmaf(wb[s], kb[(size_t)s * H], acc);
    ctx[(size_t)b * H + h] = acc;
}

// ---------------------------------------------------------------------------
// Launch
// ---------------------------------------------------------------------------
extern "C" void kernel_launch(void* const* d_in, const int* in_sizes, int n_in,
                              void* d_out, int out_size, void* d_ws, size_t ws_size,
                              hipStream_t stream) {
    const float* query = (const float*)d_in[0];   // [B,1,H]
    const float* keys  = (const float*)d_in[1];   // [B,S,H]
    const float* Wa_w  = (const float*)d_in[2];   // [H,H]
    const float* Wa_b  = (const float*)d_in[3];   // [H]
    const float* Ua_w  = (const float*)d_in[4];   // [H,H]
    const float* Ua_b  = (const float*)d_in[5];   // [H]
    const float* Va_w  = (const float*)d_in[6];   // [H,1]
    const float* Va_b  = (const float*)d_in[7];   // [1]

    float* out_ctx = (float*)d_out;                 // [B*H] context
    float* out_w   = (float*)d_out + (size_t)B * H; // [B*S] weights

    // Workspace layout (32B-aligned):
    //   [0, 128KB)        qp      : B*H fp32
    //   [128KB, 384KB)    scores  : B*S fp32
    //   [384KB, ~2.4MB)   Ua16T   : H*H bf16 (transposed)
    char* ws = (char*)d_ws;
    float*          qp     = (float*)(ws + 0);
    float*          scores = (float*)(ws + 131072);
    unsigned short* ua16   = (unsigned short*)(ws + 393216);

    cvt_transpose_kernel<<<(H * H) / 256, 256, 0, stream>>>(Ua_w, ua16);
    qproj_kernel<<<(B * H) / 256, 256, 0, stream>>>(query, Wa_w, Wa_b, qp);
    score_kernel<<<(B * S) / TM, 256, 0, stream>>>(keys, ua16, qp, Ua_b,
                                                   Va_w, Va_b, scores);
    softmax_kernel<<<B, 256, 0, stream>>>(scores, out_w);
    context_kernel<<<B * (H / 256), 256, 0, stream>>>(keys, out_w, out_ctx);
}
